// RNN_CRF_23313082483632
// MI455X (gfx1250) — matrix operand order repaired and checked by reading, not compiled, not run on hardware
//
#include <hip/hip_runtime.h>
#include <hip/hip_bf16.h>

// ---- problem dims -------------------------------------------------------
#define T_   256
#define B_   64
#define TB   16384          // T_*B_
#define H_   512
#define NT_  14

typedef __attribute__((ext_vector_type(16))) __bf16 v16bf;
typedef __attribute__((ext_vector_type(8)))  __bf16 v8bf;
typedef __attribute__((ext_vector_type(8)))  float  v8f;

__device__ __forceinline__ __bf16 f2bf(float f) {
  unsigned u = __builtin_bit_cast(unsigned, f);
  unsigned r = u + 0x7FFFu + ((u >> 16) & 1u);     // round-to-nearest-even
  unsigned short h = (unsigned short)(r >> 16);
  return __builtin_bit_cast(__bf16, h);
}
__device__ __forceinline__ float bf2f(__bf16 b) {
  unsigned short h = __builtin_bit_cast(unsigned short, b);
  return __builtin_bit_cast(float, (unsigned)h << 16);
}

// =========================================================================
// Big-GEMM path:  C[z] = A[z] @ W[z]^T + bias[z]   (bf16 in, bf16 out)
// A: [M,K] row-major (lda), W: [N,K] row-major (ldb).
// grid = (N/256, M/16, Z), block = 128 (4 waves).
// Block computes a 16(M) x 256(N) tile; the 16xKC A panel is staged into
// LDS once per chunk with GLOBAL_LOAD_ASYNC_TO_LDS_B128 (ASYNCcnt) and
// shared by all 4 waves; each wave runs 4 accumulators (16x64 strip), so
// each A fragment feeds 4 v_wmma_f32_16x16x32_bf16 ops.
// =========================================================================
__global__ void wmma_gemm_lds(const __bf16* __restrict__ A, long aZ, int lda,
                              const __bf16* __restrict__ W, long wZ, int ldb,
                              const float*  __restrict__ bias, long biasZ,
                              __bf16* __restrict__ Cb, long cbZ, int ldc,
                              int K) {
  __shared__ __bf16 As[16 * 64];                   // 2 KB A panel (16 x 64)

  const int tid  = threadIdx.x;
  const int lane = tid & 31;
  const int wv   = tid >> 5;
  const int z    = blockIdx.z;
  const int row0 = blockIdx.y << 4;
  const int colW = (blockIdx.x * 4 + wv) << 6;     // 64 cols per wave

  A += (size_t)z * aZ;
  W += (size_t)z * wZ;

  const int m  = lane & 15;
  const int hA = (lane >> 4) << 4;   // A frag: lanes 0-15 -> K 0..15; 16-31 -> K 16..31
  const int hB = (lane >> 4) << 4;   // B frag: lanes 0-15 -> K 0..15; 16-31 -> K 16..31

  // async-copy source for this thread: row r, 8-col group c8 of the 16x64 panel
  const int r  = tid >> 3;
  const int c8 = (tid & 7) << 3;
  const __bf16* gA = A + (size_t)(row0 + r) * lda + c8;
  const unsigned ldsDst = (unsigned)(size_t)(&As[r * 64 + c8]); // LDS byte offset

  const __bf16* brow0 = W + (size_t)(colW + m) * ldb + hB;
  const long    bcolStride = (long)16 * ldb;       // next 16-col tile of W

  v8f acc[4];
#pragma unroll
  for (int t = 0; t < 4; ++t)
#pragma unroll
    for (int i = 0; i < 8; ++i) acc[t][i] = 0.f;

  const __bf16* Asm = &As[m * 64];

  for (int k0 = 0; k0 < K; k0 += 64) {
    // stage 16x64 A panel into LDS (one b128 async copy per thread)
    asm volatile("global_load_async_to_lds_b128 %0, %1, off"
                 :: "v"(ldsDst), "v"(gA + k0) : "memory");
    asm volatile("s_wait_asynccnt 0x0" ::: "memory");
    __syncthreads();
#pragma unroll
    for (int kk = 0; kk < 64; kk += 32) {
      v16bf a;
      {
        v8bf lo = *(const v8bf*)(Asm + kk + hA);
        v8bf hi = *(const v8bf*)(Asm + kk + hA + 8);
#pragma unroll
        for (int i = 0; i < 8; ++i) { a[i] = lo[i]; a[i + 8] = hi[i]; }
      }
#pragma unroll
      for (int t = 0; t < 4; ++t) {
        v16bf b = *(const v16bf*)(brow0 + (long)t * bcolStride + k0 + kk);
        acc[t] = __builtin_amdgcn_wmma_f32_16x16x32_bf16(false, a, false, b,
                                                         (short)0, acc[t], false, false);
      }
    }
    __syncthreads();                               // protect As before next stage
  }

  const int n  = lane & 15;
  const int mh = (lane >> 4) << 3;
#pragma unroll
  for (int t = 0; t < 4; ++t) {
    const int col = colW + t * 16 + n;
    float bv = bias ? bias[(size_t)z * biasZ + col] : 0.f;
#pragma unroll
    for (int v = 0; v < 8; ++v) {
      float val = acc[t][v] + bv;
      Cb[(size_t)z * cbZ + (size_t)(row0 + mh + v) * ldc + col] = f2bf(val);
    }
  }
}

// =========================================================================
// Small-M GEMM (recurrent step):  C[z] = A[z] @ W[z]^T + bias[z]
// One 16x16 tile per wave, operands streamed from L2 (weights resident),
// no barriers -> minimal latency for the 256-step serial scan.
// grid = (N/64, M/16, Z), block = 128.
// =========================================================================
__global__ void wmma_gemm(const __bf16* __restrict__ A, long aZ, int lda,
                          const __bf16* __restrict__ W, long wZ, int ldb,
                          const float*  __restrict__ bias, long biasZ,
                          float* Cf, long cfZ,
                          __bf16* Cb, long cbZ, int ldc,
                          int K) {
  const int lane = threadIdx.x & 31;
  const int wv   = threadIdx.x >> 5;
  const int z    = blockIdx.z;
  const int row0 = blockIdx.y << 4;
  const int col0 = (blockIdx.x * 4 + wv) << 4;

  A += (size_t)z * aZ;
  W += (size_t)z * wZ;

  const int m  = lane & 15;
  const int hA = (lane >> 4) << 4;
  const int hB = (lane >> 4) << 4;

  const __bf16* arow = A + (size_t)(row0 + m) * lda;
  const __bf16* brow = W + (size_t)(col0 + m) * ldb + hB;

  v8f acc = {0.f, 0.f, 0.f, 0.f, 0.f, 0.f, 0.f, 0.f};
  for (int k = 0; k < K; k += 32) {
    __builtin_prefetch(brow + k + 32, 0, 1);
    v16bf a;
    {
      v8bf lo = *(const v8bf*)(arow + k + hA);
      v8bf hi = *(const v8bf*)(arow + k + hA + 8);
#pragma unroll
      for (int i = 0; i < 8; ++i) { a[i] = lo[i]; a[i + 8] = hi[i]; }
    }
    v16bf b = *(const v16bf*)(brow + k);
    acc = __builtin_amdgcn_wmma_f32_16x16x32_bf16(false, a, false, b,
                                                  (short)0, acc, false, false);
  }

  const int n  = lane & 15;
  const int mh = (lane >> 4) << 3;
  const int col = col0 + n;
  float bv = bias ? bias[(size_t)z * biasZ + col] : 0.f;
#pragma unroll
  for (int v = 0; v < 8; ++v) {
    float val = acc[v] + bv;
    size_t idx = (size_t)(row0 + mh + v) * ldc + col;
    if (Cf) Cf[(size_t)z * cfZ + idx] = val;
    if (Cb) Cb[(size_t)z * cbZ + idx] = f2bf(val);
  }
}

// ---- weight convert + pad (f32 [N,K] -> bf16 [N,Kp], zeros in pad) ------
__global__ void cvt_pad_kernel(const float* __restrict__ src, __bf16* __restrict__ dst,
                               int K, int Kp, long total) {
  long tid = (long)blockIdx.x * blockDim.x + threadIdx.x;
  if (tid >= total) return;
  long n = tid / Kp;
  int  k = (int)(tid - n * Kp);
  dst[tid] = (k < K) ? f2bf(src[n * K + k]) : f2bf(0.f);
}

// pack two f32 bias vectors contiguously: dst = [a ; b]
__global__ void pack2f(const float* __restrict__ a, const float* __restrict__ b,
                       float* __restrict__ dst, int n) {
  int i = blockIdx.x * blockDim.x + threadIdx.x;
  if (i < n) dst[i] = a[i];
  else if (i < 2 * n) dst[i] = b[i - n];
}

// features [B,T,188] f32 -> Ax [t*64+b, 192] bf16 (zero pad 188..191)
__global__ void feat_kernel(const float* __restrict__ feat, __bf16* __restrict__ Ax) {
  long tid = (long)blockIdx.x * blockDim.x + threadIdx.x;   // TB*192
  long row = tid / 192;
  int  k   = (int)(tid - row * 192);
  int  t   = (int)(row >> 6);
  int  b   = (int)(row & 63);
  Ax[tid] = (k < 188) ? f2bf(feat[((size_t)b * T_ + t) * 188 + k]) : f2bf(0.f);
}

// GRU gate update (PyTorch r,z,n order); gridDim.z = direction
__global__ void gru_gate(const __bf16* __restrict__ xp, long xpZ,
                         const float* __restrict__ hp, long hpZ,
                         const float* __restrict__ hOld, float* __restrict__ hNew, long hsZ,
                         __bf16* __restrict__ hNewB, long hsbZ,
                         __bf16* __restrict__ outB, int outLd, int step) {
  int e   = blockIdx.x * blockDim.x + threadIdx.x;   // 0..B_*H_-1
  int dir = blockIdx.z;
  int b   = e >> 9;
  int j   = e & 511;
  int t   = dir ? (T_ - 1 - step) : step;
  size_t row = (size_t)t * B_ + b;
  const __bf16* xr = xp + (size_t)dir * xpZ + row * 1536;
  const float*  hr = hp + (size_t)dir * hpZ + (size_t)b * 1536;
  float ho = hOld[(size_t)dir * hsZ + (size_t)b * 512 + j];
  float r  = 1.f / (1.f + __expf(-(bf2f(xr[j])        + hr[j])));
  float zg = 1.f / (1.f + __expf(-(bf2f(xr[512 + j])  + hr[512 + j])));
  float nn = tanhf(bf2f(xr[1024 + j]) + r * hr[1024 + j]);
  float h  = (1.f - zg) * nn + zg * ho;
  hNew [(size_t)dir * hsZ  + (size_t)b * 512 + j] = h;
  hNewB[(size_t)dir * hsbZ + (size_t)b * 512 + j] = f2bf(h);
  outB[row * outLd + (size_t)dir * 512 + j] = f2bf(h);
}

// k/v projections of NE table (batch-independent): [14,1024] each
__global__ void kv_kernel(const float* __restrict__ ne,
                          const float* __restrict__ kw, const float* __restrict__ kb,
                          const float* __restrict__ vw, const float* __restrict__ vb,
                          float* __restrict__ kmat, float* __restrict__ vmat) {
  int tid = blockIdx.x * blockDim.x + threadIdx.x;  // 14*1024
  if (tid >= NT_ * 1024) return;
  int s = tid >> 10, o = tid & 1023;
  float aK = kb[o], aV = vb[o];
#pragma unroll
  for (int j = 0; j < NT_; ++j) {
    float nv = ne[s * NT_ + j];
    aK += nv * kw[o * NT_ + j];
    aV += nv * vw[o * NT_ + j];
  }
  kmat[tid] = aK;
  vmat[tid] = aV;
}

// multi-head attention, thread = (row=t*64+b, head)
__global__ void attn_kernel(const __bf16* __restrict__ q,
                            const float* __restrict__ kmat,
                            const float* __restrict__ vmat,
                            __bf16* __restrict__ ao) {
  int tid = blockIdx.x * blockDim.x + threadIdx.x;  // TB*8
  int row = tid >> 3;
  int h   = tid & 7;
  const __bf16* qr = q + (size_t)row * 1024 + h * 128;
  const float scale = 0.08838834764831845f;         // 1/sqrt(128)
  float sc[NT_];
  float mx = -1e30f;
#pragma unroll
  for (int s = 0; s < NT_; ++s) {
    const float* kr = kmat + s * 1024 + h * 128;
    float acc = 0.f;
    for (int d = 0; d < 128; ++d) acc += bf2f(qr[d]) * kr[d];
    acc *= scale;
    sc[s] = acc;
    mx = fmaxf(mx, acc);
  }
  float sum = 0.f;
#pragma unroll
  for (int s = 0; s < NT_; ++s) { sc[s] = __expf(sc[s] - mx); sum += sc[s]; }
  float inv = 1.f / sum;
  __bf16* orow = ao + (size_t)row * 1024 + h * 128;
  for (int d = 0; d < 128; ++d) {
    float acc = 0.f;
#pragma unroll
    for (int s = 0; s < NT_; ++s) acc += sc[s] * vmat[s * 1024 + h * 128 + d];
    orow[d] = f2bf(acc * inv);
  }
}

// dec_lin[row,j] = dec[row,:1024]·lin_w[j] + lin_b[j]
__global__ void declin_kernel(const __bf16* __restrict__ dec,
                              const float* __restrict__ lw, const float* __restrict__ lb,
                              float* __restrict__ dl) {
  int tid = blockIdx.x * blockDim.x + threadIdx.x;  // TB*14
  int row = tid / NT_, j = tid % NT_;
  const __bf16* dr = dec + (size_t)row * 1024;
  const float*  wr = lw + j * 1024;
  float acc = lb[j];
  for (int c = 0; c < 1024; ++c) acc += bf2f(dr[c]) * wr[c];
  dl[tid] = acc;
}

// logits[b,t,s] = (dec_lin[t*64+b,:]·ne_table[s,:]) / sqrt(14)
__global__ void logits_kernel(const float* __restrict__ dl,
                              const float* __restrict__ ne,
                              float* __restrict__ lg) {
  int tid = blockIdx.x * blockDim.x + threadIdx.x;  // B*T*14
  int s  = tid % NT_;
  int bt = tid / NT_;
  int t  = bt % T_;
  int b  = bt / T_;
  size_t row = (size_t)t * B_ + b;
  float acc = 0.f;
#pragma unroll
  for (int j = 0; j < NT_; ++j) acc += dl[row * NT_ + j] * ne[s * NT_ + j];
  lg[tid] = acc * 0.2672612419124244f;              // 1/sqrt(14)
}

// CRF NLL: one block of 64 threads, one batch element per thread
__global__ void crf_kernel(const float* __restrict__ logits, const int* __restrict__ labels,
                           const float* __restrict__ start, const float* __restrict__ endv,
                           const float* __restrict__ trans, float* __restrict__ out) {
  int b = threadIdx.x;
  const float* lg  = logits + (size_t)b * T_ * NT_;
  const int*   lab = labels + (size_t)b * T_;

  float num = start[lab[0]];
  for (int t = 0; t < T_; ++t) num += lg[t * NT_ + lab[t]];
  for (int t = 0; t < T_ - 1; ++t) num += trans[lab[t] * NT_ + lab[t + 1]];
  num += endv[lab[T_ - 1]];

  float a[NT_];
#pragma unroll
  for (int s = 0; s < NT_; ++s) a[s] = start[s] + lg[s];
  for (int t = 1; t < T_; ++t) {
    float na[NT_];
#pragma unroll
    for (int sp = 0; sp < NT_; ++sp) {
      float m = -1e30f;
#pragma unroll
      for (int s = 0; s < NT_; ++s) m = fmaxf(m, a[s] + trans[s * NT_ + sp]);
      float sum = 0.f;
#pragma unroll
      for (int s = 0; s < NT_; ++s) sum += __expf(a[s] + trans[s * NT_ + sp] - m);
      na[sp] = m + __logf(sum) + lg[t * NT_ + sp];
    }
#pragma unroll
    for (int s = 0; s < NT_; ++s) a[s] = na[s];
  }
  float m = -1e30f;
#pragma unroll
  for (int s = 0; s < NT_; ++s) m = fmaxf(m, a[s] + endv[s]);
  float sum = 0.f;
#pragma unroll
  for (int s = 0; s < NT_; ++s) sum += __expf(a[s] + endv[s] - m);
  float den = m + __logf(sum);

  __shared__ float red[B_];
  red[b] = num - den;
  __syncthreads();
  if (b == 0) {
    float s = 0.f;
    for (int i = 0; i < B_; ++i) s += red[i];
    out[0] = -s / (float)B_;
  }
}

// ---- host orchestration -------------------------------------------------
extern "C" void kernel_launch(void* const* d_in, const int* in_sizes, int n_in,
                              void* d_out, int out_size, void* d_ws, size_t ws_size,
                              hipStream_t stream) {
  (void)in_sizes; (void)n_in; (void)out_size; (void)ws_size;
  const float* features = (const float*)d_in[0];
  const int*   labels   = (const int*)d_in[1];
  const float* ne       = (const float*)d_in[2];
  const float* g1_wih_f = (const float*)d_in[3];
  const float* g1_whh_f = (const float*)d_in[4];
  const float* g1_bih_f = (const float*)d_in[5];
  const float* g1_bhh_f = (const float*)d_in[6];
  const float* g1_wih_b = (const float*)d_in[7];
  const float* g1_whh_b = (const float*)d_in[8];
  const float* g1_bih_b = (const float*)d_in[9];
  const float* g1_bhh_b = (const float*)d_in[10];
  const float* g2_wih_f = (const float*)d_in[11];
  const float* g2_whh_f = (const float*)d_in[12];
  const float* g2_bih_f = (const float*)d_in[13];
  const float* g2_bhh_f = (const float*)d_in[14];
  const float* g2_wih_b = (const float*)d_in[15];
  const float* g2_whh_b = (const float*)d_in[16];
  const float* g2_bih_b = (const float*)d_in[17];
  const float* g2_bhh_b = (const float*)d_in[18];
  const float* q_w   = (const float*)d_in[19];
  const float* q_b   = (const float*)d_in[20];
  const float* k_w   = (const float*)d_in[21];
  const float* k_b   = (const float*)d_in[22];
  const float* v_w   = (const float*)d_in[23];
  const float* v_b   = (const float*)d_in[24];
  const float* out_w = (const float*)d_in[25];
  const float* out_b = (const float*)d_in[26];
  const float* lin_w = (const float*)d_in[27];
  const float* lin_b = (const float*)d_in[28];
  const float* crf_s = (const float*)d_in[29];
  const float* crf_e = (const float*)d_in[30];
  const float* crf_t = (const float*)d_in[31];

  char* base = (char*)d_ws;
  size_t off = 0;
  auto alloc = [&](size_t bytes) -> void* {
    void* p = base + off;
    off += (bytes + 255) & ~(size_t)255;
    return p;
  };

  __bf16* Ax    = (__bf16*)alloc((size_t)TB * 192 * 2);
  __bf16* W1IH  = (__bf16*)alloc((size_t)2 * 1536 * 192 * 2);
  __bf16* W1HH  = (__bf16*)alloc((size_t)2 * 1536 * 512 * 2);
  __bf16* W2IH  = (__bf16*)alloc((size_t)2 * 1536 * 2048 * 2);
  __bf16* W2HH  = (__bf16*)alloc((size_t)2 * 1536 * 512 * 2);
  __bf16* QW    = (__bf16*)alloc((size_t)1024 * 1024 * 2);
  __bf16* OW    = (__bf16*)alloc((size_t)1024 * 1024 * 2);
  float*  B1IH  = (float*)alloc(2 * 1536 * 4);
  float*  B1HH  = (float*)alloc(2 * 1536 * 4);
  float*  B2IH  = (float*)alloc(2 * 1536 * 4);
  float*  B2HH  = (float*)alloc(2 * 1536 * 4);
  __bf16* XP    = (__bf16*)alloc((size_t)2 * TB * 1536 * 2);     // shared gru1/gru2
  __bf16* DECIN = (__bf16*)alloc((size_t)TB * 2048 * 2);         // enc | attn-out
  __bf16* QB    = (__bf16*)alloc((size_t)TB * 1024 * 2);
  __bf16* AOB   = (__bf16*)alloc((size_t)TB * 1024 * 2);
  __bf16* DECB  = (__bf16*)alloc((size_t)TB * 1024 * 2);
  float*  HP    = (float*)alloc((size_t)2 * B_ * 1536 * 4);
  float*  HSF   = (float*)alloc((size_t)2 * 2 * B_ * 512 * 4);   // [dir][parity][B][H]
  __bf16* HSB   = (__bf16*)alloc((size_t)2 * 2 * B_ * 512 * 2);
  float*  KMAT  = (float*)alloc(NT_ * 1024 * 4);
  float*  VMAT  = (float*)alloc(NT_ * 1024 * 4);
  float*  DLIN  = (float*)alloc((size_t)TB * NT_ * 4);
  float*  LG    = (float*)alloc((size_t)B_ * T_ * NT_ * 4);

  // ---- weight / bias preparation ----
  auto cvt = [&](const float* src, __bf16* dst, int K, int Kp, long N) {
    long total = N * (long)Kp;
    cvt_pad_kernel<<<dim3((unsigned)((total + 255) / 256)), 256, 0, stream>>>(src, dst, K, Kp, total);
  };
  cvt(g1_wih_f, W1IH,                       188, 192, 1536);
  cvt(g1_wih_b, W1IH + (size_t)1536 * 192,  188, 192, 1536);
  cvt(g1_whh_f, W1HH,                       512, 512, 1536);
  cvt(g1_whh_b, W1HH + (size_t)1536 * 512,  512, 512, 1536);
  cvt(g2_wih_f, W2IH,                       2048, 2048, 1536);
  cvt(g2_wih_b, W2IH + (size_t)1536 * 2048, 2048, 2048, 1536);
  cvt(g2_whh_f, W2HH,                       512, 512, 1536);
  cvt(g2_whh_b, W2HH + (size_t)1536 * 512,  512, 512, 1536);
  cvt(q_w,   QW, 1024, 1024, 1024);
  cvt(out_w, OW, 1024, 1024, 1024);
  pack2f<<<dim3(12), 256, 0, stream>>>(g1_bih_f, g1_bih_b, B1IH, 1536);
  pack2f<<<dim3(12), 256, 0, stream>>>(g1_bhh_f, g1_bhh_b, B1HH, 1536);
  pack2f<<<dim3(12), 256, 0, stream>>>(g2_bih_f, g2_bih_b, B2IH, 1536);
  pack2f<<<dim3(12), 256, 0, stream>>>(g2_bhh_f, g2_bhh_b, B2HH, 1536);

  feat_kernel<<<dim3(TB * 192 / 256), 256, 0, stream>>>(features, Ax);

  // ---- scan helper: 256 steps x {recurrent WMMA GEMM, gate update} ------
  auto run_scan = [&](const __bf16* whh, const float* bhh, __bf16* outB, int outLd) {
    hipMemsetAsync(HSF, 0, (size_t)2 * 2 * B_ * 512 * 4, stream);
    hipMemsetAsync(HSB, 0, (size_t)2 * 2 * B_ * 512 * 2, stream);
    for (int s = 0; s < T_; ++s) {
      int p = s & 1;
      // hp[dir] = h_bf[dir][p] @ whh[dir]^T + bhh[dir]
      wmma_gemm<<<dim3(1536 / 64, B_ / 16, 2), 128, 0, stream>>>(
          HSB + (size_t)p * B_ * 512, (long)2 * B_ * 512, 512,
          whh, (long)1536 * 512, 512,
          bhh, 1536L,
          HP, (long)B_ * 1536,
          nullptr, 0L, 1536,
          512);
      gru_gate<<<dim3(B_ * 512 / 256, 1, 2), 256, 0, stream>>>(
          XP, (long)TB * 1536,
          HP, (long)B_ * 1536,
          HSF + (size_t)p * B_ * 512, HSF + (size_t)(1 - p) * B_ * 512, (long)2 * B_ * 512,
          HSB + (size_t)(1 - p) * B_ * 512, (long)2 * B_ * 512,
          outB, outLd, s);
    }
  };

  // ---- GRU1: input projection (both dirs batched in z), then scan -------
  wmma_gemm_lds<<<dim3(1536 / 256, TB / 16, 2), 128, 0, stream>>>(
      Ax, 0L, 192,
      W1IH, (long)1536 * 192, 192,
      B1IH, 1536L,
      XP, (long)TB * 1536, 1536,
      192);
  run_scan(W1HH, B1HH, DECIN, 2048);     // enc -> DECIN cols [0,1024)

  // ---- attention --------------------------------------------------------
  kv_kernel<<<dim3(NT_ * 1024 / 256), 256, 0, stream>>>(ne, k_w, k_b, v_w, v_b, KMAT, VMAT);
  wmma_gemm_lds<<<dim3(1024 / 256, TB / 16, 1), 128, 0, stream>>>(  // q = enc @ q_w^T + q_b
      DECIN, 0L, 2048,
      QW, 0L, 1024,
      q_b, 0L,
      QB, 0L, 1024,
      1024);
  attn_kernel<<<dim3(TB * 8 / 256), 256, 0, stream>>>(QB, KMAT, VMAT, AOB);
  wmma_gemm_lds<<<dim3(1024 / 256, TB / 16, 1), 128, 0, stream>>>(  // out proj -> DECIN[:,1024:]
      AOB, 0L, 1024,
      OW, 0L, 1024,
      out_b, 0L,
      DECIN + 1024, 0L, 2048,
      1024);

  // ---- GRU2: input projection over dec_in (K=2048), then scan -----------
  wmma_gemm_lds<<<dim3(1536 / 256, TB / 16, 2), 128, 0, stream>>>(
      DECIN, 0L, 2048,
      W2IH, (long)1536 * 2048, 2048,
      B2IH, 1536L,
      XP, (long)TB * 1536, 1536,
      2048);
  run_scan(W2HH, B2HH, DECB, 1024);

  // ---- linear + logits + CRF -------------------------------------------
  declin_kernel<<<dim3(TB * NT_ / 256), 256, 0, stream>>>(DECB, lin_w, lin_b, DLIN);
  logits_kernel<<<dim3(B_ * T_ * NT_ / 256), 256, 0, stream>>>(DLIN, ne, LG);
  crf_kernel<<<dim3(1), B_, 0, stream>>>(LG, labels, crf_s, crf_e, crf_t, (float*)d_out);
}
